// CoordGen_20633022890636
// MI455X (gfx1250) — matrix operand order repaired and checked by reading, not compiled, not run on hardware
//
#include <hip/hip_runtime.h>
#include <hip/hip_bf16.h>
#include <math.h>

#define B_    256
#define A_    32
#define N_    8192
#define E_    196608
#define H_    128
#define LAT_  128
#define FH_   256
#define NRBF_ 16
#define NSTEP_ 50

typedef __attribute__((ext_vector_type(16))) _Float16 v16h;
typedef __attribute__((ext_vector_type(8)))  _Float16 v8h;
typedef __attribute__((ext_vector_type(8)))  float    v8f;

// ---------------------------------------------------------------------------
// Kernel 1: lattice matrices + inverses, one thread per batch
// ---------------------------------------------------------------------------
__global__ void k_latinv(const float* __restrict__ lengths,
                         const float* __restrict__ angles,
                         float* __restrict__ lat, float* __restrict__ inv) {
    int b = blockIdx.x * blockDim.x + threadIdx.x;
    if (b >= B_) return;
    const float D2R = 0.017453292519943295f;
    float al = angles[b*3+0]*D2R, be = angles[b*3+1]*D2R, ga = angles[b*3+2]*D2R;
    float ca = cosf(al), cb = cosf(be), cg = cosf(ga);
    float sa = sinf(al), sb = sinf(be);
    float val = (ca*cb - cg) / (sa*sb);
    val = fminf(1.0f, fmaxf(-1.0f, val));
    float gs = acosf(val);
    float a  = lengths[b*3+0], bl = lengths[b*3+1], c = lengths[b*3+2];
    float M[9];
    M[0] = a*sb;              M[1] = 0.0f;            M[2] = a*cb;
    M[3] = -bl*sa*cosf(gs);   M[4] = bl*sa*sinf(gs);  M[5] = bl*ca;
    M[6] = 0.0f;              M[7] = 0.0f;            M[8] = c;
    #pragma unroll
    for (int i = 0; i < 9; ++i) lat[b*9+i] = M[i];
    float det = M[0]*(M[4]*M[8]-M[5]*M[7]) - M[1]*(M[3]*M[8]-M[5]*M[6])
              + M[2]*(M[3]*M[7]-M[4]*M[6]);
    float id = 1.0f / det;
    inv[b*9+0] = (M[4]*M[8]-M[5]*M[7])*id;
    inv[b*9+1] = (M[2]*M[7]-M[1]*M[8])*id;
    inv[b*9+2] = (M[1]*M[5]-M[2]*M[4])*id;
    inv[b*9+3] = (M[5]*M[6]-M[3]*M[8])*id;
    inv[b*9+4] = (M[0]*M[8]-M[2]*M[6])*id;
    inv[b*9+5] = (M[2]*M[3]-M[0]*M[5])*id;
    inv[b*9+6] = (M[3]*M[7]-M[4]*M[6])*id;
    inv[b*9+7] = (M[1]*M[6]-M[0]*M[7])*id;
    inv[b*9+8] = (M[0]*M[4]-M[1]*M[3])*id;
}

// ---------------------------------------------------------------------------
// Kernel 2: per-node coordinate pipeline -> pert, gt_aligned (cartesian)
// ---------------------------------------------------------------------------
__global__ void k_node(const float* __restrict__ gtf, const float* __restrict__ noise,
                       const float* __restrict__ sigmas, const int* __restrict__ tsteps,
                       const int* __restrict__ batch,
                       const float* __restrict__ lat, const float* __restrict__ inv,
                       float* __restrict__ pert, float* __restrict__ gtal) {
    int n = blockIdx.x * blockDim.x + threadIdx.x;
    if (n >= N_) return;
    int b = batch[n];
    const float* L  = lat + b*9;
    const float* IV = inv + b*9;
    float f0 = gtf[n*3+0], f1 = gtf[n*3+1], f2 = gtf[n*3+2];
    float c0 = f0*L[0] + f1*L[3] + f2*L[6];
    float c1 = f0*L[1] + f1*L[4] + f2*L[7];
    float c2 = f0*L[2] + f1*L[5] + f2*L[8];
    float sig = sigmas[tsteps[b]];
    float p0 = c0 + sig*noise[n*3+0];
    float p1 = c1 + sig*noise[n*3+1];
    float p2 = c2 + sig*noise[n*3+2];
    float q0 = p0*IV[0] + p1*IV[3] + p2*IV[6];
    float q1 = p0*IV[1] + p1*IV[4] + p2*IV[7];
    float q2 = p0*IV[2] + p1*IV[5] + p2*IV[8];
    q0 -= floorf(q0); q1 -= floorf(q1); q2 -= floorf(q2);
    pert[n*3+0] = q0*L[0] + q1*L[3] + q2*L[6];
    pert[n*3+1] = q0*L[1] + q1*L[4] + q2*L[7];
    pert[n*3+2] = q0*L[2] + q1*L[5] + q2*L[8];
    float g0 = c0*IV[0] + c1*IV[3] + c2*IV[6];
    float g1 = c0*IV[1] + c1*IV[4] + c2*IV[7];
    float g2 = c0*IV[2] + c1*IV[5] + c2*IV[8];
    float d0 = g0 - q0; d0 -= rintf(d0);
    float d1 = g1 - q1; d1 -= rintf(d1);
    float d2 = g2 - q2; d2 -= rintf(d2);
    float u0 = q0 + d0, u1 = q1 + d1, u2 = q2 + d2;
    gtal[n*3+0] = u0*L[0] + u1*L[3] + u2*L[6];
    gtal[n*3+1] = u0*L[1] + u1*L[4] + u2*L[7];
    gtal[n*3+2] = u0*L[2] + u1*L[5] + u2*L[8];
}

// ---------------------------------------------------------------------------
// Kernel 3: repack f32 weights [K][N] into f16 WMMA B-fragment tile order.
// Tile t = kt*NTl + nt holds 32 lanes x 16 halves; lane L: column = nt*16+(L&15),
// K = kt*32 + (L>>4)*8 + (e&7) + (e&8)*2  (matches V_WMMA_F32_16X16X32_F16 B layout).
// ---------------------------------------------------------------------------
__global__ void k_pack(const float* __restrict__ W, _Float16* __restrict__ dst,
                       int Kreal, int Ncols, int KT, int NTl) {
    int t = blockIdx.x * blockDim.x + threadIdx.x;
    int lane = t & 31;
    int tile = t >> 5;
    if (tile >= KT * NTl) return;
    int kt = tile / NTl, nt = tile % NTl;
    int col = nt*16 + (lane & 15);
    int k0  = kt*32 + ((lane >> 4) << 3);
    v16h v;
    #pragma unroll
    for (int e = 0; e < 16; ++e) {
        int k = k0 + (e & 7) + ((e & 8) << 1);
        v[e] = (k < Kreal) ? (_Float16)W[(size_t)k * Ncols + col] : (_Float16)0.0f;
    }
    *(v16h*)(dst + (size_t)tile * 512 + lane * 16) = v;
}

// ---------------------------------------------------------------------------
// Kernel 4: zero node accumulators + output scalar
// ---------------------------------------------------------------------------
__global__ void k_zero(float* __restrict__ acc, float* __restrict__ out) {
    int i = blockIdx.x * blockDim.x + threadIdx.x;
    if (i < 2 * N_ * 3) acc[i] = 0.0f;
    if (i == 0) out[0] = 0.0f;
}

// ---------------------------------------------------------------------------
// WMMA GEMM tile helper: one wave computes rows [mbase, mbase+16) of the
// 128-row LDS tile against all NTL 16-col weight tiles; bias + SiLU + f16
// store into smOut (pitch 256). A-frag load follows the 16-bit A layout:
// lanes 0-15 take K {kb..kb+7, kb+16..kb+23}, lanes 16-31 the +8 halves.
// ---------------------------------------------------------------------------
template<int PITCH_A, int KT, int NTL>
__device__ __forceinline__ void gemm_silu(const _Float16* __restrict__ smIn,
                                          const _Float16* __restrict__ Wp,
                                          const float* __restrict__ bias,
                                          _Float16* __restrict__ smOut,
                                          int mbase, int lane) {
    const int row  = mbase + (lane & 15);
    const int khof = (lane >> 4) << 3;   // +0 or +8
    const int moff = (lane < 16) ? 0 : 8;
    for (int nh = 0; nh < NTL / 8; ++nh) {
        v8f acc[8] = {};
        for (int kt = 0; kt < KT; ++kt) {
            const _Float16* pa = smIn + row * PITCH_A + kt * 32 + khof;
            v8h lo = *(const v8h*)(pa);
            v8h hi = *(const v8h*)(pa + 16);
            v16h a;
            #pragma unroll
            for (int i = 0; i < 8; ++i) { a[i] = lo[i]; a[i + 8] = hi[i]; }
            #pragma unroll
            for (int nt2 = 0; nt2 < 8; ++nt2) {
                int nt = nh * 8 + nt2;
                const v16h bfr = *(const v16h*)(Wp + (size_t)(kt * NTL + nt) * 512 + lane * 16);
                acc[nt2] = __builtin_amdgcn_wmma_f32_16x16x32_f16(
                    false, a, false, bfr, (short)0, acc[nt2], false, false);
            }
        }
        #pragma unroll
        for (int nt2 = 0; nt2 < 8; ++nt2) {
            int ncol = (nh * 8 + nt2) * 16 + (lane & 15);
            float bv = bias[ncol];
            #pragma unroll
            for (int r = 0; r < 8; ++r) {
                float x = acc[nt2][r] + bv;
                float s = x / (1.0f + __expf(-x));           // SiLU
                smOut[(mbase + moff + r) * 256 + ncol] = (_Float16)s;
            }
        }
    }
}

// ---------------------------------------------------------------------------
// Kernel 5: fused per-edge MLP (128 edges / workgroup, 8 waves)
// ---------------------------------------------------------------------------
__global__ __launch_bounds__(256)
void k_edge(const float* __restrict__ latents, const float* __restrict__ atom_emb,
            const int* __restrict__ atom_types, const int* __restrict__ batch,
            const int* __restrict__ eidx, const int* __restrict__ tji,
            const float* __restrict__ lat,
            const float* __restrict__ pert, const float* __restrict__ gtal,
            const _Float16* __restrict__ Wbp, const float* __restrict__ bb,
            const _Float16* __restrict__ W1p, const float* __restrict__ b1,
            const _Float16* __restrict__ W2p, const float* __restrict__ b2,
            const float* __restrict__ W3, const float* __restrict__ b3,
            float* __restrict__ node_s, float* __restrict__ node_g) {
    extern __shared__ char smem_raw[];
    _Float16* smA   = (_Float16*)(smem_raw);             // 128*288 f16 = 73728 B
    _Float16* smB   = (_Float16*)(smem_raw + 73728);     // 128*256 f16 = 65536 B
    float*    g_dv  = (float*)(smem_raw + 139264);       // 128*3 f32
    float*    g_dist= (float*)(smem_raw + 140800);       // 128 f32
    float*    g_gts = (float*)(smem_raw + 141312);       // 128 f32
    float*    g_sc  = (float*)(smem_raw + 141824);       // 128 f32
    int*      g_ii  = (int*)  (smem_raw + 142336);       // 128 i32

    const int tid  = threadIdx.x;
    const int wave = tid >> 5;
    const int lane = tid & 31;
    const int ebase = blockIdx.x * 128;

    // ---------- Phase 0: assemble X0 (128x288 f16) + geometry ----------
    {
        int r  = tid & 127;
        int pp = tid >> 7;
        int e  = ebase + r;
        int jj = eidx[e];
        int ii = eidx[E_ + e];
        if (pp == 0) {
            const float* em = atom_emb + (size_t)atom_types[jj] * H_;
            for (int c = 0; c < H_; c += 4) {
                float4 v = *(const float4*)(em + c);
                smA[r*288 + c + 0] = (_Float16)v.x;
                smA[r*288 + c + 1] = (_Float16)v.y;
                smA[r*288 + c + 2] = (_Float16)v.z;
                smA[r*288 + c + 3] = (_Float16)v.w;
            }
            const float* lp = latents + (size_t)batch[jj] * LAT_;
            for (int c = 0; c < LAT_; c += 4) {
                float4 v = *(const float4*)(lp + c);
                smB[r*256 + 128 + c + 0] = (_Float16)v.x;
                smB[r*256 + 128 + c + 1] = (_Float16)v.y;
                smB[r*256 + 128 + c + 2] = (_Float16)v.z;
                smB[r*256 + 128 + c + 3] = (_Float16)v.w;
            }
        } else {
            const float* em = atom_emb + (size_t)atom_types[ii] * H_;
            for (int c = 0; c < H_; c += 4) {
                float4 v = *(const float4*)(em + c);
                smA[r*288 + 128 + c + 0] = (_Float16)v.x;
                smA[r*288 + 128 + c + 1] = (_Float16)v.y;
                smA[r*288 + 128 + c + 2] = (_Float16)v.z;
                smA[r*288 + 128 + c + 3] = (_Float16)v.w;
            }
            const float* L = lat + (size_t)batch[ii] * 9;
            float t0 = (float)tji[e*3+0], t1 = (float)tji[e*3+1], t2 = (float)tji[e*3+2];
            float dv[3], dg[3];
            float dd2 = 0.0f, gg2 = 0.0f;
            #pragma unroll
            for (int j = 0; j < 3; ++j) {
                float off = t0*L[j] + t1*L[3+j] + t2*L[6+j];
                dv[j] = pert[ii*3+j] - pert[jj*3+j] + off;
                dg[j] = gtal[ii*3+j] - gtal[jj*3+j] + off;
                dd2 += dv[j]*dv[j];
                gg2 += dg[j]*dg[j];
                g_dv[r*3+j] = dv[j];
            }
            float dist = sqrtf(dd2), gd = sqrtf(gg2);
            g_dist[r] = dist;
            g_gts[r]  = gd - dist;
            g_ii[r]   = ii;
            #pragma unroll
            for (int k = 0; k < NRBF_; ++k) {
                float ck = 0.46666667f * (float)k;          // linspace(0,7,16)
                float d  = dist - ck;
                smA[r*288 + 256 + k] = (_Float16)__expf(-d*d * 2.6122449f); // /width2
            }
            #pragma unroll
            for (int k = 0; k < 16; ++k) smA[r*288 + 272 + k] = (_Float16)0.0f;
        }
    }
    __syncthreads();

    const int mbase = wave * 16;
    // ---------- Phase 1: h = silu(X0 @ Wb + bb)  -> smB cols [0,128) ----------
    gemm_silu<288, 9, 8>(smA, Wbp, bb, smB, mbase, lane);
    __syncthreads();
    // ---------- Phase 2: s1 = silu(feat @ W1 + b1) -> smA (pitch 256) ----------
    gemm_silu<256, 8, 16>(smB, W1p, b1, smA, mbase, lane);
    __syncthreads();
    // ---------- Phase 3: s2 = silu(s1 @ W2 + b2) -> smB ----------
    gemm_silu<256, 8, 16>(smA, W2p, b2, smB, mbase, lane);
    __syncthreads();

    // ---------- Phase 4: scores = s2 @ W3 + b3 ----------
    if (tid < 128) {
        const _Float16* row = smB + tid * 256;
        float acc = 0.0f;
        for (int k = 0; k < 256; k += 4) {
            acc += (float)row[k+0]*W3[k+0] + (float)row[k+1]*W3[k+1]
                 + (float)row[k+2]*W3[k+2] + (float)row[k+3]*W3[k+3];
        }
        g_sc[tid] = acc + b3[0];
    }
    __syncthreads();

    // ---------- Phase 5: pair-average (cnt==2) + scatter to nodes ----------
    if (tid < 64) {
        int r0 = tid * 2, r1 = r0 + 1;
        float s0 = g_sc[r0]  / g_dist[r0];
        float s1 = g_sc[r1]  / g_dist[r1];
        float t0 = g_gts[r0] / g_dist[r0];
        float t1 = g_gts[r1] / g_dist[r1];
        int   ii = g_ii[r0];
        #pragma unroll
        for (int c = 0; c < 3; ++c) {
            float vec = 0.5f * (s0 * g_dv[r0*3+c] + s1 * g_dv[r1*3+c]);
            float gv  = 0.5f * (t0 * g_dv[r0*3+c] + t1 * g_dv[r1*3+c]);
            atomicAdd(&node_s[ii*3+c], vec);
            atomicAdd(&node_g[ii*3+c], gv);
        }
    }
}

// ---------------------------------------------------------------------------
// Kernel 6: final loss reduction. gcnt is uniformly 32, so the result is the
// plain mean over all N*3 elements of (node_s - node_g/sn)^2.
// ---------------------------------------------------------------------------
__global__ void k_loss(const float* __restrict__ node_s, const float* __restrict__ node_g,
                       const float* __restrict__ snorms, const int* __restrict__ tsteps,
                       const int* __restrict__ batch, float* __restrict__ out) {
    int n = blockIdx.x * blockDim.x + threadIdx.x;
    if (n >= N_) return;
    float sn = snorms[tsteps[batch[n]]];
    float acc = 0.0f;
    #pragma unroll
    for (int c = 0; c < 3; ++c) {
        float d = node_s[n*3+c] - node_g[n*3+c] / sn;
        acc += d * d;
    }
    atomicAdd(out, acc * (1.0f / 24576.0f));   // 1 / (A * B * 3)
}

// ---------------------------------------------------------------------------
extern "C" void kernel_launch(void* const* d_in, const int* in_sizes, int n_in,
                              void* d_out, int out_size, void* d_ws, size_t ws_size,
                              hipStream_t stream) {
    (void)in_sizes; (void)n_in; (void)out_size; (void)ws_size;
    const float* latents = (const float*)d_in[0];
    const float* gtf     = (const float*)d_in[1];
    const float* lengths = (const float*)d_in[2];
    const float* angles  = (const float*)d_in[3];
    const float* noise   = (const float*)d_in[4];
    const float* sigmas  = (const float*)d_in[5];
    const float* snorms  = (const float*)d_in[6];
    const float* aemb    = (const float*)d_in[7];
    const float* Wb      = (const float*)d_in[8];
    const float* bb      = (const float*)d_in[9];
    const float* W1      = (const float*)d_in[10];
    const float* b1      = (const float*)d_in[11];
    const float* W2      = (const float*)d_in[12];
    const float* b2      = (const float*)d_in[13];
    const float* W3      = (const float*)d_in[14];
    const float* b3      = (const float*)d_in[15];
    const int* atype  = (const int*)d_in[16];
    const int* batch  = (const int*)d_in[17];
    const int* eidx   = (const int*)d_in[18];
    const int* tji    = (const int*)d_in[19];
    const int* tsteps = (const int*)d_in[20];

    char* ws = (char*)d_ws;
    float*    lat    = (float*)(ws + 0);        // B*9
    float*    inv    = (float*)(ws + 9216);     // B*9
    float*    pert   = (float*)(ws + 18432);    // N*3
    float*    gtal   = (float*)(ws + 116736);   // N*3
    float*    node_s = (float*)(ws + 215040);   // N*3
    float*    node_g = (float*)(ws + 313344);   // N*3 (contiguous with node_s)
    _Float16* Wbp    = (_Float16*)(ws + 411648); // 72 tiles * 512 f16
    _Float16* W1p    = (_Float16*)(ws + 485376); // 128 tiles * 512 f16
    _Float16* W2p    = (_Float16*)(ws + 616448); // 128 tiles * 512 f16
    float* out = (float*)d_out;

    k_latinv<<<1, 256, 0, stream>>>(lengths, angles, lat, inv);
    k_node<<<N_/256, 256, 0, stream>>>(gtf, noise, sigmas, tsteps, batch, lat, inv, pert, gtal);
    k_pack<<<(72*32 + 255)/256, 256, 0, stream>>>(Wb, Wbp, 272, 128, 9, 8);
    k_pack<<<(128*32 + 255)/256, 256, 0, stream>>>(W1, W1p, 256, 256, 8, 16);
    k_pack<<<(128*32 + 255)/256, 256, 0, stream>>>(W2, W2p, 256, 256, 8, 16);
    k_zero<<<(2*N_*3 + 255)/256, 256, 0, stream>>>(node_s, out);

    const size_t smem_bytes = 142848;   // smA + smB + geometry/scores
    k_edge<<<E_/128, 256, smem_bytes, stream>>>(
        latents, aemb, atype, batch, eidx, tji, lat, pert, gtal,
        Wbp, bb, W1p, b1, W2p, b2, W3, b3, node_s, node_g);

    k_loss<<<N_/256, 256, 0, stream>>>(node_s, node_g, snorms, tsteps, batch, out);
}
